// DotAttackHead_62130996904141
// MI455X (gfx1250) — compile-verified, matrix operand order checked by reading, not run on hardware
//
#include <hip/hip_runtime.h>
#include <hip/hip_bf16.h>
#include <stdint.h>

// ---- problem dims (fixed by reference) ----
#define BATCH 64
#define PNU   1024
#define PNE   1024
#define KDIM  512   // UFEAT_DIM
#define EDIM  256   // EFEAT_DIM

// LDS layout (dynamic shared, overlapped regions):
//   [0      , 16640)  ufeatS : 16 x 520 bf16 (stage A input tile)   -- dead after stage A
//   [16640  , 25088)  projS  : 16 x 264 bf16 (proj tile)            -- dead after stage B
//   [0      , 65792)  outS   : 16 x 1028 f32 (prob tile, overlaps both above)
//   [65792  , 66304)  redmax : 16 x 8 f32
//   [66304  , 66816)  redsum : 16 x 8 f32
#define UFEATS_OFF 0
#define PROJS_OFF  16640
#define OUTS_STRIDE 1028
#define REDMAX_OFF 65792
#define REDSUM_OFF 66304
#define SMEM_BYTES 66816

// ws layout: [0,4) scale | [256, 256+256KB) Wb bf16 | [512KB, 512KB+32MB) Eb bf16
#define WS_WB_OFF  256
#define WS_EB_OFF  (512 * 1024)
#define WS_EB_BYTES ((size_t)BATCH * PNE * EDIM * 2)

typedef __attribute__((ext_vector_type(16))) __bf16 v16bf;
typedef __attribute__((ext_vector_type(8)))  float  v8f;

union BFPack {
    v16bf v;
    unsigned int u[8];
};

__device__ __forceinline__ unsigned short f2bf(float f) {
    unsigned int x = __builtin_bit_cast(unsigned int, f);
    x += 0x7FFFu + ((x >> 16) & 1u);          // round-to-nearest-even
    return (unsigned short)(x >> 16);
}
__device__ __forceinline__ unsigned int packbf(float lo, float hi) {
    return (unsigned int)f2bf(lo) | ((unsigned int)f2bf(hi) << 16);
}

// ---- kernel 0: scale = g / ||v||_F ----
__global__ __launch_bounds__(256) void wnorm_kernel(const float* __restrict__ v,
                                                    const float* __restrict__ g,
                                                    float* __restrict__ scale) {
    __shared__ float sbuf[256];
    const int tid = threadIdx.x;
    float s = 0.f;
    for (int i = tid; i < EDIM * KDIM; i += 256) {
        float x = v[i];
        s += x * x;
    }
    sbuf[tid] = s;
    __syncthreads();
    for (int off = 128; off > 0; off >>= 1) {
        if (tid < off) sbuf[tid] += sbuf[tid + off];
        __syncthreads();
    }
    if (tid == 0) scale[0] = g[0] * rsqrtf(sbuf[0]);
}

// ---- kernel 1: Wb = bf16(scale * v) ----
__global__ __launch_bounds__(256) void wconv_kernel(const float* __restrict__ v,
                                                    const float* __restrict__ scale,
                                                    unsigned short* __restrict__ Wb) {
    const int i = blockIdx.x * 256 + threadIdx.x;
    Wb[i] = f2bf(v[i] * scale[0]);
}

// ---- kernel 2: Eb = bf16(efeat), removes per-tile f32->bf16 VALU in main loop ----
__global__ __launch_bounds__(256) void econv_kernel(const float* __restrict__ e,
                                                    unsigned short* __restrict__ Eb) {
    const int i = blockIdx.x * 256 + threadIdx.x;
    Eb[i] = f2bf(e[i]);
}

// ---- main fused kernel: proj GEMM + logit GEMM + masked softmax ----
template <bool EB16>
__global__ __launch_bounds__(256) void attack_head_kernel(
        const float* __restrict__ ufeat,
        const float* __restrict__ efeat,
        const unsigned short* __restrict__ Eb,
        const long long* __restrict__ num_enemy,
        const float* __restrict__ bias,
        const unsigned short* __restrict__ Wb,
        float* __restrict__ out) {
    extern __shared__ __align__(16) unsigned char smem[];
    unsigned short* ufeatS = (unsigned short*)(smem + UFEATS_OFF);
    unsigned short* projS  = (unsigned short*)(smem + PROJS_OFF);
    float*          outS   = (float*)(smem);
    float*          redmax = (float*)(smem + REDMAX_OFF);   // [16][8]
    float*          redsum = (float*)(smem + REDSUM_OFF);   // [16][8]

    const int tid    = threadIdx.x;
    const int lane   = tid & 31;
    const int wid    = tid >> 5;
    const int lane16 = lane & 15;
    const int half   = lane >> 4;
    const int bid    = blockIdx.y;
    const int u0     = blockIdx.x * 16;
    const int kboff  = half * 8;   // A/B layout: lanes 16-31 own K+8..K+15 / K+24..K+31

    // ---- stage 0: stage ufeat tile [16 x 512] into LDS as bf16 (vectorized) ----
    {
        const float* src = ufeat + ((size_t)bid * PNU + u0) * KDIM;
        for (int i = tid * 4; i < 16 * KDIM; i += 256 * 4) {
            const int r = i >> 9;            // KDIM == 512
            const int c = i & (KDIM - 1);
            float4 f = *(const float4*)(src + r * KDIM + c);
            uint2 p;
            p.x = packbf(f.x, f.y);
            p.y = packbf(f.z, f.w);
            *(uint2*)(ufeatS + r * 520 + c) = p;
        }
    }
    __syncthreads();

    // ---- stage A: proj = relu(U @ W^T + b), bf16 into projS ----
    #pragma unroll
    for (int ei = 0; ei < 2; ++ei) {
        const int et   = wid * 2 + ei;        // this wave's E-column tile
        const int ecol = et * 16 + lane16;    // W row = output column
        v8f acc = {0.f, 0.f, 0.f, 0.f, 0.f, 0.f, 0.f, 0.f};
        for (int k = 0; k < KDIM; k += 32) {
            const int kb = k + kboff;
            BFPack a;
            {
                uint4 t0 = *(const uint4*)(ufeatS + lane16 * 520 + kb);
                uint4 t1 = *(const uint4*)(ufeatS + lane16 * 520 + kb + 16);
                a.u[0] = t0.x; a.u[1] = t0.y; a.u[2] = t0.z; a.u[3] = t0.w;
                a.u[4] = t1.x; a.u[5] = t1.y; a.u[6] = t1.z; a.u[7] = t1.w;
            }
            BFPack bm;
            {
                uint4 t0 = *(const uint4*)(Wb + (size_t)ecol * KDIM + kb);
                uint4 t1 = *(const uint4*)(Wb + (size_t)ecol * KDIM + kb + 16);
                bm.u[0] = t0.x; bm.u[1] = t0.y; bm.u[2] = t0.z; bm.u[3] = t0.w;
                bm.u[4] = t1.x; bm.u[5] = t1.y; bm.u[6] = t1.z; bm.u[7] = t1.w;
            }
            acc = __builtin_amdgcn_wmma_f32_16x16x32_bf16(false, a.v, false, bm.v,
                                                          (short)0, acc, false, false);
        }
        const float bv = bias[ecol];
        #pragma unroll
        for (int j = 0; j < 8; ++j) {
            float p = acc[j] + bv;
            p = p > 0.f ? p : 0.f;                      // ReLU
            projS[(j + half * 8) * 264 + ecol] = f2bf(p);
        }
    }
    __syncthreads();

    // ---- stage B: logits for this wave's 128-wide N chunk ----
    const int n0 = wid * 128;
    const long long ne = num_enemy[bid];
    v8f acc[8];
    const v8f vzero = {0.f, 0.f, 0.f, 0.f, 0.f, 0.f, 0.f, 0.f};
    #pragma unroll
    for (int nt = 0; nt < 8; ++nt) acc[nt] = vzero;

    for (int e = 0; e < EDIM; e += 32) {
        const int kb = e + kboff;
        BFPack a;
        {
            uint4 t0 = *(const uint4*)(projS + lane16 * 264 + kb);
            uint4 t1 = *(const uint4*)(projS + lane16 * 264 + kb + 16);
            a.u[0] = t0.x; a.u[1] = t0.y; a.u[2] = t0.z; a.u[3] = t0.w;
            a.u[4] = t1.x; a.u[5] = t1.y; a.u[6] = t1.z; a.u[7] = t1.w;
        }
        #pragma unroll
        for (int nt = 0; nt < 8; ++nt) {
            const int n = n0 + nt * 16 + lane16;        // efeat row
            BFPack bm;
            if (EB16) {
                const unsigned short* pe = Eb + ((size_t)bid * PNE + n) * EDIM + kb;
                uint4 t0 = *(const uint4*)(pe);
                uint4 t1 = *(const uint4*)(pe + 16);
                bm.u[0] = t0.x; bm.u[1] = t0.y; bm.u[2] = t0.z; bm.u[3] = t0.w;
                bm.u[4] = t1.x; bm.u[5] = t1.y; bm.u[6] = t1.z; bm.u[7] = t1.w;
            } else {
                const float* pe = efeat + ((size_t)bid * PNE + n) * EDIM + kb;
                float4 f0 = *(const float4*)(pe);
                float4 f1 = *(const float4*)(pe + 4);
                float4 f2 = *(const float4*)(pe + 16);
                float4 f3 = *(const float4*)(pe + 20);
                bm.u[0] = packbf(f0.x, f0.y); bm.u[1] = packbf(f0.z, f0.w);
                bm.u[2] = packbf(f1.x, f1.y); bm.u[3] = packbf(f1.z, f1.w);
                bm.u[4] = packbf(f2.x, f2.y); bm.u[5] = packbf(f2.z, f2.w);
                bm.u[6] = packbf(f3.x, f3.y); bm.u[7] = packbf(f3.z, f3.w);
            }
            acc[nt] = __builtin_amdgcn_wmma_f32_16x16x32_bf16(false, a.v, false, bm.v,
                                                              (short)0, acc[nt], false, false);
        }
    }

    // ---- mask + scale; per-row running max over this wave's chunk ----
    float partmax[8];
    #pragma unroll
    for (int j = 0; j < 8; ++j) partmax[j] = -3.4e38f;
    #pragma unroll
    for (int nt = 0; nt < 8; ++nt) {
        const int n = n0 + nt * 16 + lane16;
        const float mb = ((long long)n < ne) ? 0.f : -1e9f;  // enemy mask bias
        #pragma unroll
        for (int j = 0; j < 8; ++j) {
            float l = acc[nt][j] * 0.0625f + mb;             // /sqrt(256)
            acc[nt][j] = l;
            partmax[j] = fmaxf(partmax[j], l);
        }
    }
    // butterfly within each 16-lane half (C layout: half selects rows 0-7 / 8-15)
    #pragma unroll
    for (int off = 1; off < 16; off <<= 1) {
        #pragma unroll
        for (int j = 0; j < 8; ++j)
            partmax[j] = fmaxf(partmax[j], __shfl_xor(partmax[j], off, 32));
    }
    if (lane16 == 0) {
        #pragma unroll
        for (int j = 0; j < 8; ++j) redmax[(j + half * 8) * 8 + wid] = partmax[j];
    }
    __syncthreads();

    float rmax[8];
    #pragma unroll
    for (int j = 0; j < 8; ++j) {
        float m = redmax[(j + half * 8) * 8];
        #pragma unroll
        for (int w = 1; w < 8; ++w) m = fmaxf(m, redmax[(j + half * 8) * 8 + w]);
        rmax[j] = m;
    }

    // ---- exp once (numerator is final since rmax is global), partial sums ----
    float partsum[8];
    #pragma unroll
    for (int j = 0; j < 8; ++j) partsum[j] = 0.f;
    #pragma unroll
    for (int nt = 0; nt < 8; ++nt) {
        #pragma unroll
        for (int j = 0; j < 8; ++j) {
            float ev = __expf(acc[nt][j] - rmax[j]);
            acc[nt][j] = ev;
            partsum[j] += ev;
        }
    }
    #pragma unroll
    for (int off = 1; off < 16; off <<= 1) {
        #pragma unroll
        for (int j = 0; j < 8; ++j)
            partsum[j] += __shfl_xor(partsum[j], off, 32);
    }
    if (lane16 == 0) {
        #pragma unroll
        for (int j = 0; j < 8; ++j) redsum[(j + half * 8) * 8 + wid] = partsum[j];
    }
    __syncthreads();

    float rinv[8];
    #pragma unroll
    for (int j = 0; j < 8; ++j) {
        float s = redsum[(j + half * 8) * 8];
        #pragma unroll
        for (int w = 1; w < 8; ++w) s += redsum[(j + half * 8) * 8 + w];
        rinv[j] = 1.f / s;
    }

    // ---- stage prob tile in LDS (stride 1028 -> halves land on bank sets +32) ----
    #pragma unroll
    for (int nt = 0; nt < 8; ++nt) {
        const int n = n0 + nt * 16 + lane16;
        #pragma unroll
        for (int j = 0; j < 8; ++j) {
            const int row = j + half * 8;
            outS[row * OUTS_STRIDE + n] = acc[nt][j] * rinv[j];
        }
    }
    __syncthreads();

    // ---- cooperative fully-coalesced writeback (512B contiguous per wave/store) ----
    {
        float* dst = out + ((size_t)bid * PNU + u0) * PNE;
        #pragma unroll
        for (int i = tid * 4; i < 16 * PNE; i += 256 * 4) {
            const int row = i >> 10;                 // PNE == 1024
            const int col = i & (PNE - 1);
            float4 val = *(const float4*)(outS + row * OUTS_STRIDE + col);
            *(float4*)(dst + (size_t)row * PNE + col) = val;
        }
    }
}

extern "C" void kernel_launch(void* const* d_in, const int* in_sizes, int n_in,
                              void* d_out, int out_size, void* d_ws, size_t ws_size,
                              hipStream_t stream) {
    const float*     ufeat     = (const float*)d_in[0];
    const float*     efeat     = (const float*)d_in[1];
    const long long* num_enemy = (const long long*)d_in[2];
    const float*     v         = (const float*)d_in[3];
    const float*     g         = (const float*)d_in[4];
    const float*     b         = (const float*)d_in[5];

    float* scale = (float*)d_ws;
    unsigned short* Wb = (unsigned short*)((char*)d_ws + WS_WB_OFF);
    unsigned short* Eb = (unsigned short*)((char*)d_ws + WS_EB_OFF);

    wnorm_kernel<<<1, 256, 0, stream>>>(v, g, scale);
    wconv_kernel<<<(EDIM * KDIM) / 256, 256, 0, stream>>>(v, scale, Wb);

    const bool useEb = ws_size >= (size_t)WS_EB_OFF + WS_EB_BYTES;
    dim3 grid(PNU / 16, BATCH);
    if (useEb) {
        econv_kernel<<<(BATCH * PNE * EDIM) / 256, 256, 0, stream>>>(efeat, Eb);
        attack_head_kernel<true><<<grid, 256, SMEM_BYTES, stream>>>(
            ufeat, efeat, Eb, num_enemy, b, Wb, (float*)d_out);
    } else {
        attack_head_kernel<false><<<grid, 256, SMEM_BYTES, stream>>>(
            ufeat, efeat, Eb, num_enemy, b, Wb, (float*)d_out);
    }
}
